// QuanGAT_87582973100319
// MI455X (gfx1250) — matrix operand order repaired and checked by reading, not compiled, not run on hardware
//
#include <hip/hip_runtime.h>
#include <cmath>

typedef __bf16 bf16;
typedef __attribute__((ext_vector_type(16))) __bf16 v16bf;
typedef __attribute__((ext_vector_type(8)))  float  v8f;

#define N_NODES 50000
#define N_EDGES 800000
#define ETOT    850000   // edges + self loops
#define IN_CH   128
#define HID     128
#define OUT_CH  64

// ---------------------------------------------------------------- utilities
__device__ inline float wave_reduce(float v) {
#pragma unroll
  for (int off = 16; off > 0; off >>= 1) v += __shfl_down(v, off, 32);
  return v;
}

__device__ inline void atomicMaxFloat(float* addr, float value) {
  if (value >= 0.f) atomicMax((int*)addr, __float_as_int(value));
  else              atomicMin((unsigned int*)addr, __float_as_uint(value));
}

__device__ inline void edge_sd(const int* __restrict__ ei, int e, int& s, int& d) {
  if (e < N_EDGES) { s = ei[e]; d = ei[N_EDGES + e]; }
  else             { s = e - N_EDGES; d = s; }
}

__global__ void fill_kernel(float* p, float v, size_t n) {
  size_t i = (size_t)blockIdx.x * blockDim.x + threadIdx.x;
  if (i < n) p[i] = v;
}

// ------------------------------------------------------ weight pre-transpose
__global__ void convert_w1t(const float* __restrict__ W1, bf16* __restrict__ W1t) {
  int t = blockIdx.x * blockDim.x + threadIdx.x;         // 512*128
  if (t >= 512 * 128) return;
  int col = t >> 7, k = t & 127;
  W1t[t] = (bf16)W1[k * 512 + col];                      // W1 is [128][512]
}
__global__ void convert_w2t(const float* __restrict__ W2, bf16* __restrict__ W2t) {
  int t = blockIdx.x * blockDim.x + threadIdx.x;         // 64*512
  if (t >= 64 * 512) return;
  int col = t >> 9, k = t & 511;
  W2t[t] = (bf16)W2[k * 64 + col];                       // W2 is [512][64]
}

// ---------------------------------------------------------------- proj GEMM
__global__ __launch_bounds__(256) void proj_kernel(
    const float* __restrict__ x, const float* __restrict__ Wp,
    const float* __restrict__ bp, float* __restrict__ hout) {
  __shared__ float sW[128 * 4];
  __shared__ float sb[4];
  int tid = threadIdx.x;
  if (tid < 512) sW[tid] = Wp[tid];
  if (tid < 4)   sb[tid] = bp[tid];
  __syncthreads();
  int wave = tid >> 5, lane = tid & 31;
  int node = blockIdx.x * 8 + wave;
  if (node >= N_NODES) return;
  float p0 = 0, p1 = 0, p2 = 0, p3 = 0;
  for (int k = lane; k < 128; k += 32) {
    float xv = x[(size_t)node * 128 + k];
    p0 += xv * sW[k * 4 + 0]; p1 += xv * sW[k * 4 + 1];
    p2 += xv * sW[k * 4 + 2]; p3 += xv * sW[k * 4 + 3];
  }
  p0 = wave_reduce(p0); p1 = wave_reduce(p1);
  p2 = wave_reduce(p2); p3 = wave_reduce(p3);
  if (lane == 0) {
    hout[node * 4 + 0] = p0 + sb[0]; hout[node * 4 + 1] = p1 + sb[1];
    hout[node * 4 + 2] = p2 + sb[2]; hout[node * 4 + 3] = p3 + sb[3];
  }
}

// ------------------------------------------------------------- quantum + FC
__device__ inline void rx_gate(float* sr, float* si, float theta, int w) {
  float sv, cv;
  __sincosf(theta * 0.5f, &sv, &cv);
  int mask = 1 << (3 - w);
#pragma unroll
  for (int i = 0; i < 16; ++i) {
    if (i & mask) continue;
    int j = i | mask;
    float ar = sr[i], ai = si[i], br = sr[j], bi = si[j];
    sr[i] = cv * ar + sv * bi;  si[i] = cv * ai - sv * br;
    sr[j] = sv * ai + cv * br;  si[j] = cv * bi - sv * ar;
  }
}

__device__ inline void cnot_gate(float* sr, float* si, int ctrl, int tgt) {
  int cm = 1 << (3 - ctrl), tm = 1 << (3 - tgt);
#pragma unroll
  for (int i = 0; i < 16; ++i) {
    if ((i & cm) && !(i & tm)) {
      int j = i | tm;
      float tr = sr[i], ti = si[i];
      sr[i] = sr[j]; si[i] = si[j];
      sr[j] = tr;    si[j] = ti;
    }
  }
}

__global__ __launch_bounds__(256) void quantum_fc_kernel(
    const float* __restrict__ hproj, const float* __restrict__ qw,
    const float* __restrict__ Wfc, const float* __restrict__ bfc,
    bf16* __restrict__ h2bf) {
  __shared__ float sW[4 * 128];
  __shared__ float sb[128];
  __shared__ float sq[12];
  int tid = threadIdx.x;
  if (tid < 512) sW[tid] = Wfc[tid];
  if (tid < 128) sb[tid] = bfc[tid];
  if (tid < 12)  sq[tid] = qw[tid];
  __syncthreads();
  int node = blockIdx.x * 256 + tid;
  if (node >= N_NODES) return;

  float sr[16], si[16];
#pragma unroll
  for (int i = 0; i < 16; ++i) { sr[i] = 0.f; si[i] = 0.f; }
  sr[0] = 1.f;
#pragma unroll
  for (int w = 0; w < 4; ++w) rx_gate(sr, si, hproj[node * 4 + w], w);
#pragma unroll
  for (int l = 0; l < 3; ++l) {
#pragma unroll
    for (int w = 0; w < 4; ++w) rx_gate(sr, si, sq[l * 4 + w], w);
#pragma unroll
    for (int w = 0; w < 4; ++w) cnot_gate(sr, si, w, (w + 1) & 3);
  }
  float qh[4];
#pragma unroll
  for (int w = 0; w < 4; ++w) {
    int mask = 1 << (3 - w);
    float a = 0.f;
#pragma unroll
    for (int i = 0; i < 16; ++i) {
      float p = sr[i] * sr[i] + si[i] * si[i];
      a += (i & mask) ? -p : p;
    }
    qh[w] = a;
  }
  bf16* dst = h2bf + (size_t)node * 128;
#pragma unroll 4
  for (int c = 0; c < 128; ++c) {
    float v = sb[c] + qh[0] * sW[c] + qh[1] * sW[128 + c] +
              qh[2] * sW[256 + c] + qh[3] * sW[384 + c];
    dst[c] = (bf16)v;
  }
}

// ------------------------------------------------------------ WMMA fragments
// A fragment 16x32 (MxK) bf16, row-major src, row stride ld.
__device__ inline v16bf load_a_frag(const bf16* __restrict__ src, int ld, int lane) {
  int m = lane & 15, hi = lane >> 4;
  v16bf a;
#pragma unroll
  for (int j = 0; j < 8; ++j) {
    int k = ((j >> 2) << 4) + (hi << 3) + ((j & 3) << 1);
    a[2 * j]     = src[m * ld + k];
    a[2 * j + 1] = src[m * ld + k + 1];
  }
  return a;
}
// B fragment 32x16 (KxN) bf16 from transposed weights wt[col][k], col stride ldk.
__device__ inline v16bf load_b_frag(const bf16* __restrict__ src, int ldk, int lane) {
  int n = lane & 15, hi = lane >> 4;
  v16bf b;
#pragma unroll
  for (int j = 0; j < 8; ++j) {
    int k = ((j >> 2) << 4) + (hi << 3) + ((j & 3) << 1);
    b[2 * j]     = src[n * ldk + k];
    b[2 * j + 1] = src[n * ldk + k + 1];
  }
  return b;
}

// ----------------------------------------------- GEMM1: [N,128]x[128,512]
__global__ __launch_bounds__(256) void gemm1_kernel(
    const bf16* __restrict__ A, const bf16* __restrict__ Wt, float* __restrict__ C) {
  __shared__ bf16 As[16 * 128];                 // 4 KB A tile
  int m0 = blockIdx.x * 16;
  int tid = threadIdx.x;
  {
    const uint4* gsrc = (const uint4*)(A + (size_t)m0 * 128);
    uint4* sdst = (uint4*)As;
    sdst[tid] = gsrc[tid];                      // 256 * 16B = 4096B
  }
  __syncthreads();
  int wave = tid >> 5, lane = tid & 31;
  v8f acc[4];
#pragma unroll
  for (int nt = 0; nt < 4; ++nt)
#pragma unroll
    for (int r = 0; r < 8; ++r) acc[nt][r] = 0.f;

  for (int kk = 0; kk < 128; kk += 32) {
    v16bf af = load_a_frag(As + kk, 128, lane);
#pragma unroll
    for (int nt = 0; nt < 4; ++nt) {
      int col0 = wave * 64 + nt * 16;
      v16bf bfr = load_b_frag(Wt + (size_t)col0 * 128 + kk, 128, lane);
      acc[nt] = __builtin_amdgcn_wmma_f32_16x16x32_bf16(
          false, af, false, bfr, (short)0, acc[nt], false, false);
    }
  }
  int n = lane & 15, hi = lane >> 4;
#pragma unroll
  for (int nt = 0; nt < 4; ++nt) {
    int col = wave * 64 + nt * 16 + n;
#pragma unroll
    for (int r = 0; r < 8; ++r)
      C[(size_t)(m0 + r + 8 * hi) * 512 + col] = acc[nt][r];
  }
}

// ----------------------------------------------- GEMM2: [N,512]x[512,64]
__global__ __launch_bounds__(128) void gemm2_kernel(
    const bf16* __restrict__ A, const bf16* __restrict__ Wt, float* __restrict__ C) {
  __shared__ bf16 As[16 * 512];                 // 16 KB A tile
  int m0 = blockIdx.x * 16;
  int tid = threadIdx.x;
  {
    const uint4* gsrc = (const uint4*)(A + (size_t)m0 * 512);
    uint4* sdst = (uint4*)As;
    for (int i = tid; i < 1024; i += 128) sdst[i] = gsrc[i];
  }
  __syncthreads();
  int wave = tid >> 5, lane = tid & 31;
  v8f acc;
#pragma unroll
  for (int r = 0; r < 8; ++r) acc[r] = 0.f;

  for (int kk = 0; kk < 512; kk += 32) {
    v16bf af  = load_a_frag(As + kk, 512, lane);
    v16bf bfr = load_b_frag(Wt + (size_t)(wave * 16) * 512 + kk, 512, lane);
    acc = __builtin_amdgcn_wmma_f32_16x16x32_bf16(
        false, af, false, bfr, (short)0, acc, false, false);
  }
  int n = lane & 15, hi = lane >> 4;
#pragma unroll
  for (int r = 0; r < 8; ++r)
    C[(size_t)(m0 + r + 8 * hi) * 64 + wave * 16 + n] = acc[r];
}

// ----------------------------------------------------- attention coefficients
__global__ __launch_bounds__(256) void attn1_kernel(
    const float* __restrict__ xl, const float* __restrict__ vsrc,
    const float* __restrict__ vdst, float* __restrict__ a_s, float* __restrict__ a_d) {
  int wid = (blockIdx.x * blockDim.x + threadIdx.x) >> 5;   // node*4 + head
  int lane = threadIdx.x & 31;
  if (wid >= N_NODES * 4) return;
  int node = wid >> 2, h = wid & 3;
  const float* row = xl + (size_t)node * 512 + h * 128;
  const float* vs = vsrc + h * 128;
  const float* vd = vdst + h * 128;
  float ps = 0.f, pd = 0.f;
  for (int c = lane; c < 128; c += 32) { float v = row[c]; ps += v * vs[c]; pd += v * vd[c]; }
  ps = wave_reduce(ps); pd = wave_reduce(pd);
  if (lane == 0) { a_s[wid] = ps; a_d[wid] = pd; }
}

__global__ __launch_bounds__(256) void attn2_kernel(
    const float* __restrict__ xl, const float* __restrict__ vsrc,
    const float* __restrict__ vdst, float* __restrict__ a_s, float* __restrict__ a_d) {
  int wid = (blockIdx.x * blockDim.x + threadIdx.x) >> 5;   // node
  int lane = threadIdx.x & 31;
  if (wid >= N_NODES) return;
  const float* row = xl + (size_t)wid * 64;
  float ps = 0.f, pd = 0.f;
  for (int c = lane; c < 64; c += 32) { float v = row[c]; ps += v * vsrc[c]; pd += v * vdst[c]; }
  ps = wave_reduce(ps); pd = wave_reduce(pd);
  if (lane == 0) { a_s[wid] = ps; a_d[wid] = pd; }
}

// ------------------------------------------------------------- edge softmax 1
__global__ __launch_bounds__(256) void edge_alpha1(
    const int* __restrict__ ei, const float* __restrict__ a_s,
    const float* __restrict__ a_d, float* __restrict__ alpha, float* __restrict__ amax) {
  int t = blockIdx.x * blockDim.x + threadIdx.x;            // e*4 + h
  if (t >= ETOT * 4) return;
  int e = t >> 2, h = t & 3;
  int s, d; edge_sd(ei, e, s, d);
  float a = a_s[s * 4 + h] + a_d[d * 4 + h];
  a = (a > 0.f) ? a : 0.2f * a;                             // leaky_relu
  alpha[t] = a;
  atomicMaxFloat(&amax[d * 4 + h], a);
}

__global__ __launch_bounds__(256) void edge_exp1(
    const int* __restrict__ ei, float* __restrict__ alpha,
    const float* __restrict__ amax, float* __restrict__ denom) {
  int t = blockIdx.x * blockDim.x + threadIdx.x;
  if (t >= ETOT * 4) return;
  int e = t >> 2, h = t & 3;
  int s, d; edge_sd(ei, e, s, d);
  float ex = __expf(alpha[t] - amax[d * 4 + h]);
  alpha[t] = ex;
  atomicAdd(&denom[d * 4 + h], ex);
}

__global__ __launch_bounds__(256) void edge_msg1(
    const int* __restrict__ ei, const float* __restrict__ ex,
    const float* __restrict__ denom, const float* __restrict__ xl,
    float* __restrict__ out) {
  int gw = (blockIdx.x * blockDim.x + threadIdx.x) >> 5;    // one wave per edge
  int lane = threadIdx.x & 31;
  if (gw >= ETOT) return;
  int s, d; edge_sd(ei, gw, s, d);
  float coef[4];
#pragma unroll
  for (int h = 0; h < 4; ++h) coef[h] = ex[(size_t)gw * 4 + h] / denom[d * 4 + h];
  const float4* srcp = (const float4*)(xl + (size_t)s * 512);
  float* dstp = out + (size_t)d * 512;
  __builtin_prefetch(srcp, 0, 1);
#pragma unroll
  for (int i = 0; i < 4; ++i) {                             // head i = channels [i*128, i*128+128)
    int c4 = i * 32 + lane;
    float4 v = srcp[c4];
    float cf = coef[i];
    int c = c4 * 4;
    atomicAdd(dstp + c + 0, cf * v.x);
    atomicAdd(dstp + c + 1, cf * v.y);
    atomicAdd(dstp + c + 2, cf * v.z);
    atomicAdd(dstp + c + 3, cf * v.w);
  }
}

__global__ __launch_bounds__(256) void bias_relu_kernel(
    const float* __restrict__ acc, const float* __restrict__ b1, bf16* __restrict__ hbf) {
  size_t idx = (size_t)blockIdx.x * blockDim.x + threadIdx.x;
  if (idx >= (size_t)N_NODES * 512) return;
  float v = acc[idx] + b1[idx & 511];
  v = v > 0.f ? v : 0.f;
  hbf[idx] = (bf16)v;
}

// ------------------------------------------------------------- edge softmax 2
__global__ __launch_bounds__(256) void edge_alpha2(
    const int* __restrict__ ei, const float* __restrict__ a_s,
    const float* __restrict__ a_d, float* __restrict__ alpha, float* __restrict__ amax) {
  int e = blockIdx.x * blockDim.x + threadIdx.x;
  if (e >= ETOT) return;
  int s, d; edge_sd(ei, e, s, d);
  float a = a_s[s] + a_d[d];
  a = (a > 0.f) ? a : 0.2f * a;
  alpha[e] = a;
  atomicMaxFloat(&amax[d], a);
}

__global__ __launch_bounds__(256) void edge_exp2(
    const int* __restrict__ ei, float* __restrict__ alpha,
    const float* __restrict__ amax, float* __restrict__ denom) {
  int e = blockIdx.x * blockDim.x + threadIdx.x;
  if (e >= ETOT) return;
  int s, d; edge_sd(ei, e, s, d);
  float ex = __expf(alpha[e] - amax[d]);
  alpha[e] = ex;
  atomicAdd(&denom[d], ex);
}

__global__ __launch_bounds__(256) void edge_msg2(
    const int* __restrict__ ei, const float* __restrict__ ex,
    const float* __restrict__ denom, const float* __restrict__ xl,
    float* __restrict__ out) {
  int gw = (blockIdx.x * blockDim.x + threadIdx.x) >> 5;
  int lane = threadIdx.x & 31;
  if (gw >= ETOT) return;
  int s, d; edge_sd(ei, gw, s, d);
  float coef = ex[gw] / denom[d];
  const float2* srcp = (const float2*)(xl + (size_t)s * 64);
  float* dstp = out + (size_t)d * 64;
  float2 v = srcp[lane];
  atomicAdd(dstp + lane * 2 + 0, coef * v.x);
  atomicAdd(dstp + lane * 2 + 1, coef * v.y);
}

__global__ __launch_bounds__(256) void bias2_kernel(float* __restrict__ out,
                                                    const float* __restrict__ b2) {
  size_t idx = (size_t)blockIdx.x * blockDim.x + threadIdx.x;
  if (idx >= (size_t)N_NODES * 64) return;
  out[idx] += b2[idx & 63];
}

// ------------------------------------------------------------------- launch
extern "C" void kernel_launch(void* const* d_in, const int* in_sizes, int n_in,
                              void* d_out, int out_size, void* d_ws, size_t ws_size,
                              hipStream_t stream) {
  (void)in_sizes; (void)n_in; (void)out_size; (void)ws_size;
  const float* x      = (const float*)d_in[0];
  const int*   ei     = (const int*)  d_in[1];
  const float* W_proj = (const float*)d_in[2];
  const float* b_proj = (const float*)d_in[3];
  const float* q_w    = (const float*)d_in[4];
  const float* W_fc   = (const float*)d_in[5];
  const float* b_fc   = (const float*)d_in[6];
  const float* W1     = (const float*)d_in[7];
  const float* a1_src = (const float*)d_in[8];
  const float* a1_dst = (const float*)d_in[9];
  const float* b1     = (const float*)d_in[10];
  const float* W2     = (const float*)d_in[11];
  const float* a2_src = (const float*)d_in[12];
  const float* a2_dst = (const float*)d_in[13];
  const float* b2     = (const float*)d_in[14];
  float* out = (float*)d_out;

  char* wp = (char*)d_ws;
  auto alloc = [&](size_t bytes) -> void* {
    void* p = (void*)wp;
    wp += (bytes + 255) & ~(size_t)255;
    return p;
  };
  float* hproj  = (float*)alloc((size_t)N_NODES * 4 * 4);
  bf16*  h2bf   = (bf16*) alloc((size_t)N_NODES * 128 * 2);   // reused as xl2 (N*64 f32, same size)
  bf16*  W1t    = (bf16*) alloc((size_t)512 * 128 * 2);
  bf16*  W2t    = (bf16*) alloc((size_t)64 * 512 * 2);
  float* xl1    = (float*)alloc((size_t)N_NODES * 512 * 4);   // reused as h1 bf16 (half size)
  float* as1    = (float*)alloc((size_t)N_NODES * 4 * 4);     // reused as as2
  float* ad1    = (float*)alloc((size_t)N_NODES * 4 * 4);     // reused as ad2
  float* amax1  = (float*)alloc((size_t)N_NODES * 4 * 4);     // reused as amax2
  float* denom1 = (float*)alloc((size_t)N_NODES * 4 * 4);     // reused as denom2
  float* alpha1 = (float*)alloc((size_t)ETOT * 4 * 4);        // reused as alpha2
  float* out1   = (float*)alloc((size_t)N_NODES * 512 * 4);
  // aliases (lifetimes are disjoint on the single stream):
  bf16*  h1bf   = (bf16*)xl1;
  float* xl2    = (float*)h2bf;
  float* as2 = as1; float* ad2 = ad1; float* amax2 = amax1;
  float* denom2 = denom1; float* alpha2 = alpha1;

  // ---- stage 0: weight conversion ----
  convert_w1t<<<(512 * 128 + 255) / 256, 256, 0, stream>>>(W1, W1t);
  convert_w2t<<<(64 * 512 + 255) / 256, 256, 0, stream>>>(W2, W2t);

  // ---- stage 1: proj + quantum + fc ----
  proj_kernel<<<N_NODES / 8, 256, 0, stream>>>(x, W_proj, b_proj, hproj);
  quantum_fc_kernel<<<(N_NODES + 255) / 256, 256, 0, stream>>>(hproj, q_w, W_fc, b_fc, h2bf);

  // ---- stage 2: GAT layer 1 ----
  gemm1_kernel<<<N_NODES / 16, 256, 0, stream>>>(h2bf, W1t, xl1);
  attn1_kernel<<<N_NODES * 4 / 8, 256, 0, stream>>>(xl1, a1_src, a1_dst, as1, ad1);
  fill_kernel<<<(N_NODES * 4 + 255) / 256, 256, 0, stream>>>(amax1, -INFINITY, (size_t)N_NODES * 4);
  fill_kernel<<<(N_NODES * 4 + 255) / 256, 256, 0, stream>>>(denom1, 0.f, (size_t)N_NODES * 4);
  fill_kernel<<<(N_NODES * 512 + 255) / 256, 256, 0, stream>>>(out1, 0.f, (size_t)N_NODES * 512);
  edge_alpha1<<<(ETOT * 4 + 255) / 256, 256, 0, stream>>>(ei, as1, ad1, alpha1, amax1);
  edge_exp1<<<(ETOT * 4 + 255) / 256, 256, 0, stream>>>(ei, alpha1, amax1, denom1);
  edge_msg1<<<(ETOT + 7) / 8, 256, 0, stream>>>(ei, alpha1, denom1, xl1, out1);
  bias_relu_kernel<<<(N_NODES * 512 + 255) / 256, 256, 0, stream>>>(out1, b1, h1bf);

  // ---- stage 3: GAT layer 2 ----
  gemm2_kernel<<<N_NODES / 16, 128, 0, stream>>>(h1bf, W2t, xl2);
  attn2_kernel<<<N_NODES / 8, 256, 0, stream>>>(xl2, a2_src, a2_dst, as2, ad2);
  fill_kernel<<<(N_NODES + 255) / 256, 256, 0, stream>>>(amax2, -INFINITY, (size_t)N_NODES);
  fill_kernel<<<(N_NODES + 255) / 256, 256, 0, stream>>>(denom2, 0.f, (size_t)N_NODES);
  fill_kernel<<<(N_NODES * 64 + 255) / 256, 256, 0, stream>>>(out, 0.f, (size_t)N_NODES * 64);
  edge_alpha2<<<(ETOT + 255) / 256, 256, 0, stream>>>(ei, as2, ad2, alpha2, amax2);
  edge_exp2<<<(ETOT + 255) / 256, 256, 0, stream>>>(ei, alpha2, amax2, denom2);
  edge_msg2<<<(ETOT + 7) / 8, 256, 0, stream>>>(ei, alpha2, denom2, xl2, out);
  bias2_kernel<<<(N_NODES * 64 + 255) / 256, 256, 0, stream>>>(out, b2);
}
